// Attention_15092515078756
// MI455X (gfx1250) — matrix operand order repaired and checked by reading, not compiled, not run on hardware
//
#include <hip/hip_runtime.h>
#include <hip/hip_bf16.h>
#include <cstdint>
#include <cstddef>

// ---------------- problem constants ----------------
constexpr int Bb = 2;      // batch
constexpr int Tt = 2048;   // seq len
constexpr int Dd = 2048;   // model dim
constexpr int Nh = 16;     // q heads
constexpr int Kh = 8;      // kv heads
constexpr int Hd = 128;    // head dim
#define SOFTCAP 50.0f

// ---------------- WMMA types ----------------
typedef __attribute__((ext_vector_type(16))) __bf16 v16bf;
typedef __attribute__((ext_vector_type(8)))  float  v8f;

struct U4 { unsigned int a, b, c, d; };
union BFrag {
    U4 u[2];
    v16bf v;
    unsigned short s[16];
};

static __device__ __forceinline__ unsigned short f2bfbits(float f) {
    __hip_bfloat16 h = __float2bfloat16(f);   // RNE, native cvt where available
    return __builtin_bit_cast(unsigned short, h);
}
static __device__ __forceinline__ __bf16 f2bf(float f) {
    return __builtin_bit_cast(__bf16, f2bfbits(f));
}
static __device__ __forceinline__ float bf2f(__bf16 h) {
    unsigned int u = ((unsigned int)__builtin_bit_cast(unsigned short, h)) << 16;
    return __builtin_bit_cast(float, u);
}
// load 16 contiguous bf16 (32B, 16B-aligned) as a fragment register block
static __device__ __forceinline__ v16bf ld16(const void* p) {
    BFrag t;
    const U4* q = (const U4*)p;
    t.u[0] = q[0];
    t.u[1] = q[1];
    return t.v;
}
static __device__ __forceinline__ v8f wmma_bf16(v16bf a, v16bf b, v8f c) {
    // D(16x16 f32) = A(16x32 bf16) * B(32x16 bf16) + C
    return __builtin_amdgcn_wmma_f32_16x16x32_bf16(false, a, false, b, (short)0, c, false, false);
}
static __device__ __forceinline__ float tanh_fast(float x) {
    float ax = fabsf(x);
    float e  = __expf(-2.0f * ax);
    float t  = (1.0f - e) / (1.0f + e);
    return copysignf(t, x);
}

// =====================================================================
// Kernel 1: fused QKV projection GEMM.
//   M = B*T = 4096 rows of x (f32), cols = N*H + 2*K*H = 4096,
//   Kdim = D = 2048.  Block = 4 waves, 64x64 output tile.
//   Weights staged to LDS transposed as bf16 (B-fragment layout).
//   Q,K stored natural [bt][head][h]; V stored TRANSPOSED [b][kk][h][t]
//   so the attention PV B-fragments are direct contiguous loads.
// =====================================================================
constexpr int WSTRIDE = 40; // 32 + 8 pad (bf16 units)

__global__ __launch_bounds__(128) void qkv_proj_kernel(
    const float* __restrict__ x, const float* __restrict__ wq,
    const float* __restrict__ wkv,
    __bf16* __restrict__ qw, __bf16* __restrict__ kw, __bf16* __restrict__ vt)
{
    __shared__ unsigned short lds_w[64 * WSTRIDE];

    const int col0 = blockIdx.x * 64;    // global GEMM column
    const int row0 = blockIdx.y * 64;    // global GEMM row (b*T+t)
    const int tid  = threadIdx.x;
    const int wave = tid >> 5;
    const int lane = tid & 31;
    const int l15  = lane & 15;
    const int hi16 = lane >> 4;          // 0 / 1
    const int kb   = 8 * hi16;           // unused (A and B frags both use kb16)
    const int kb16 = 16 * hi16;          // B-frag K base
    const int m0   = row0 + wave * 16;

    // weight segment select (block-uniform; tiles never straddle q/k/v)
    const float* wsrc;
    size_t wbase;
    int hbase, mode;                      // mode: 0=q 1=k 2=v
    if (col0 < Nh * Hd) {
        int n = col0 >> 7; hbase = col0 & 127;
        wsrc = wq;  wbase = (size_t)n * Dd * Hd; mode = 0;
    } else if (col0 < Nh * Hd + Kh * Hd) {
        int c2 = col0 - Nh * Hd; int n = c2 >> 7; hbase = c2 & 127;
        wsrc = wkv; wbase = (size_t)n * Dd * Hd; mode = 1;
    } else {
        int c2 = col0 - (Nh * Hd + Kh * Hd); int n = c2 >> 7; hbase = c2 & 127;
        wsrc = wkv; wbase = ((size_t)Kh + n) * Dd * Hd; mode = 2;
    }

    v8f acc[4];
#pragma unroll
    for (int i = 0; i < 4; ++i) acc[i] = v8f{0.f,0.f,0.f,0.f,0.f,0.f,0.f,0.f};

    const int dr = tid >> 2;           // 0..31  (K row to stage)
    const int cq = (tid & 3) * 16;     // 0/16/32/48 (col base to stage)

    for (int d0 = 0; d0 < Dd; d0 += 32) {
        __syncthreads();  // previous iter's fragment reads finished
        // ---- stage W[d0..d0+32) x [col0..col0+64) transposed to LDS bf16
        const float* wp = wsrc + wbase + (size_t)(d0 + dr) * Hd + hbase + cq;
#pragma unroll
        for (int j4 = 0; j4 < 4; ++j4) {
            float4 w4 = *(const float4*)(wp + j4 * 4);
            lds_w[(cq + j4 * 4 + 0) * WSTRIDE + dr] = f2bfbits(w4.x);
            lds_w[(cq + j4 * 4 + 1) * WSTRIDE + dr] = f2bfbits(w4.y);
            lds_w[(cq + j4 * 4 + 2) * WSTRIDE + dr] = f2bfbits(w4.z);
            lds_w[(cq + j4 * 4 + 3) * WSTRIDE + dr] = f2bfbits(w4.w);
        }
        __syncthreads();

        // ---- A fragment (x rows, f32 -> bf16), k = 16*hi16 + s, contiguous like B
        const float* xrow = x + (size_t)(m0 + l15) * Dd + d0;
        BFrag ta;
#pragma unroll
        for (int j = 0; j < 8; ++j) {
            ta.s[j]     = f2bfbits(xrow[kb16 + j]);
            ta.s[8 + j] = f2bfbits(xrow[kb16 + 8 + j]);
        }
        v16bf af = ta.v;

#pragma unroll
        for (int ct = 0; ct < 4; ++ct) {
            v16bf bf = ld16(&lds_w[(ct * 16 + l15) * WSTRIDE + kb16]);
            acc[ct] = wmma_bf16(af, bf, acc[ct]);
        }
    }

    // ---- store: C/D layout is (lane = column, VGPR r = row r + 8*hi16)
    if (mode < 2) {
        __bf16* dst   = (mode == 0) ? qw : kw;
        const int rstride = (mode == 0) ? Nh * Hd : Kh * Hd;
        const int cofs    = (mode == 0) ? col0 : col0 - Nh * Hd;
#pragma unroll
        for (int ct = 0; ct < 4; ++ct) {
            int col = cofs + ct * 16 + l15;
#pragma unroll
            for (int r = 0; r < 8; ++r) {
                int row = m0 + r + 8 * hi16;
                dst[(size_t)row * rstride + col] = f2bf(acc[ct][r]);
            }
        }
    } else {
        // V transposed: vt[((b*Kh + kk)*Hd + h)*Tt + t]; 8 rows are 8
        // consecutive t values for this lane -> one packed 16B store.
        const int c2    = col0 - (Nh * Hd + Kh * Hd);
        const int kkv   = c2 >> 7;                 // block-uniform
        const int hb_v  = c2 & 127;
        const int bidx  = m0 >> 11;                // rows per batch = Tt
        const int t0    = (m0 & (Tt - 1)) + 8 * hi16;
#pragma unroll
        for (int ct = 0; ct < 4; ++ct) {
            int h = hb_v + ct * 16 + l15;
            size_t base = ((size_t)(bidx * Kh + kkv) * Hd + h) * Tt + t0;
            U4 pk;
            pk.a = (unsigned)f2bfbits(acc[ct][0]) | ((unsigned)f2bfbits(acc[ct][1]) << 16);
            pk.b = (unsigned)f2bfbits(acc[ct][2]) | ((unsigned)f2bfbits(acc[ct][3]) << 16);
            pk.c = (unsigned)f2bfbits(acc[ct][4]) | ((unsigned)f2bfbits(acc[ct][5]) << 16);
            pk.d = (unsigned)f2bfbits(acc[ct][6]) | ((unsigned)f2bfbits(acc[ct][7]) << 16);
            *(U4*)(vt + base) = pk;
        }
    }
}

// =====================================================================
// Kernel 2: RoPE on q (plus H^-1/2 scale) and k, in place, bf16.
// =====================================================================
__global__ __launch_bounds__(256) void rope_kernel(
    __bf16* __restrict__ qw, __bf16* __restrict__ kw)
{
    const long long idx = (long long)blockIdx.x * 256 + threadIdx.x;
    const int  i    = (int)(idx & 63);           // 0..H/2-1
    long long  r    = idx >> 6;
    const int  head = (int)(r % (Nh + Kh));      // 0..23
    const long long bt = r / (Nh + Kh);          // b*T + t
    const int  t    = (int)(bt % Tt);

    const float frac = (float)i * (1.0f / 64.0f);          // 2i/H
    const float ts   = __expf(frac * 9.210340371976184f);  // 10000^frac
    const float ang  = (float)t / ts;
    float sn, cs;
    __sincosf(ang, &sn, &cs);

    __bf16* base;
    float scale;
    if (head < Nh) { base = qw + (bt * Nh + head) * Hd;        scale = 0.08838834764831845f; }
    else           { base = kw + (bt * Kh + (head - Nh)) * Hd; scale = 1.0f; }

    float f1 = bf2f(base[i]);
    float f2 = bf2f(base[i + 64]);
    base[i]      = f2bf((f1 * cs - f2 * sn) * scale);
    base[i + 64] = f2bf((f2 * cs + f1 * sn) * scale);
}

// =====================================================================
// Kernel 3: attention core. One wave per (b, head n, 16-row q tile).
//   Max-free streaming softmax (softcap bounds logits to +-50).
//   Phase 1: QK^T via WMMA -> tanh softcap -> mask -> exp -> LDS
//   Phase 2: P (LDS, f32->bf16) x V-transposed (direct global loads).
// =====================================================================
constexpr int PSTRIDE = 516;  // 512 + 4 pad floats (row offset -> +32 banks)

__global__ __launch_bounds__(32) void attn_kernel(
    const __bf16* __restrict__ qw, const __bf16* __restrict__ kw,
    const __bf16* __restrict__ vt, __bf16* __restrict__ enc)
{
    __shared__ float lds_p[16 * PSTRIDE];   // exp(logits), one 512-wide s-chunk

    const int bid = blockIdx.x;
    const int qt  = bid & (Tt / 16 - 1);
    const int n   = (bid >> 7) & (Nh - 1);
    const int b   = bid >> 11;
    const int lane = threadIdx.x;
    const int l15  = lane & 15;
    const int hi16 = lane >> 4;
    const int kb   = 8 * hi16;
    const int kb16 = 16 * hi16;
    const int qbase = qt * 16;
    const int qend  = qbase + 16;
    const int kk    = n >> 1;               // GQA: G = N/K = 2

    // ---- Q fragments resident in registers (A layout: row = l15)
    const __bf16* qrow = qw + ((size_t)(b * Tt + qbase + l15) * Nh + n) * Hd;
    v16bf aq[4];
#pragma unroll
    for (int hc = 0; hc < 4; ++hc) {
        BFrag t;
        t.u[0] = *(const U4*)(qrow + hc * 32 + kb16);
        t.u[1] = *(const U4*)(qrow + hc * 32 + kb16 + 8);
        aq[hc] = t.v;
    }

    const __bf16* vhead = vt + (size_t)(b * Kh + kk) * Hd * Tt;

    v8f oacc[8];
#pragma unroll
    for (int i = 0; i < 8; ++i) oacc[i] = v8f{0.f,0.f,0.f,0.f,0.f,0.f,0.f,0.f};
    float rowsum = 0.f;
    const int trow = qbase + 8 * hi16;      // + r gives this lane's C/D row

    for (int c0 = 0; c0 < qend; c0 += 512) {
        int hiW = qend - c0; if (hiW > 512) hiW = 512;
        const int W32  = ((hiW + 31) >> 5) << 5;  // pad to 32 (extra tile fully masked)
        const int nt16 = W32 >> 4;

        // -------- phase 1: logits -> softcap -> mask -> exp -> LDS --------
        for (int st = 0; st < nt16; ++st) {
            const int sb   = c0 + st * 16;
            const int scol = sb + l15;
            const int scl  = scol < Tt ? scol : Tt - 1;
            const __bf16* krow = kw + ((size_t)(b * Tt + scl) * Kh + kk) * Hd;
            if (st + 1 < nt16) {   // look ahead: next K tile (global_prefetch_b8)
                int sp = scol + 16; sp = sp < Tt ? sp : Tt - 1;
                __builtin_prefetch(kw + ((size_t)(b * Tt + sp) * Kh + kk) * Hd, 0, 0);
            }

            v8f acc = v8f{0.f,0.f,0.f,0.f,0.f,0.f,0.f,0.f};
#pragma unroll
            for (int hc = 0; hc < 4; ++hc) {
                v16bf bk = ld16(krow + hc * 32 + kb16);  // B layout: col=l15, contig h
                acc = wmma_bf16(aq[hc], bk, acc);
            }
#pragma unroll
            for (int r = 0; r < 8; ++r) {
                float lg = SOFTCAP * tanh_fast(acc[r] * (1.0f / SOFTCAP));
                int   tg = trow + r;
                float p  = (scol <= tg) ? __expf(lg) : 0.0f;  // max-free: |lg|<=50
                lds_p[(8 * hi16 + r) * PSTRIDE + st * 16 + l15] = p;
            }
        }

        // -------- row sums over this chunk (lane pair per row) --------
        {
            float s = 0.f;
            for (int j = hi16; j < W32; j += 2)
                s += lds_p[l15 * PSTRIDE + j];
            s += __shfl_xor(s, 16, 32);
            rowsum += s;
        }

        // -------- phase 2: P x V (V already transposed: direct loads) ----
        const int ns32 = W32 >> 5;
        for (int s32 = 0; s32 < ns32; ++s32) {
            const int sb = c0 + s32 * 32;
            if (s32 + 1 < ns32)    // look ahead: 32 lanes prefetch every 4th h row (0,4,..,124)
                __builtin_prefetch(vhead + (size_t)(lane * 4) * Tt + sb + 32, 0, 0);

            // A fragment from P (row = l15), f32 -> bf16 on the fly
            const float* prow = &lds_p[l15 * PSTRIDE + s32 * 32];
            BFrag ta;
#pragma unroll
            for (int j = 0; j < 8; ++j) {
                ta.s[j]     = f2bfbits(prow[kb16 + j]);
                ta.s[8 + j] = f2bfbits(prow[kb16 + 8 + j]);
            }
            v16bf ap = ta.v;
#pragma unroll
            for (int ht = 0; ht < 8; ++ht) {
                // B frag: col h = ht*16+l15, 16 contiguous s at sb+kb16
                v16bf bv = ld16(vhead + (size_t)(ht * 16 + l15) * Tt + sb + kb16);
                oacc[ht] = wmma_bf16(ap, bv, oacc[ht]);
            }
        }
    }

    // ---- normalize by row sum and store encoded (bf16)
#pragma unroll
    for (int ht = 0; ht < 8; ++ht) {
#pragma unroll
        for (int r = 0; r < 8; ++r) {
            int   lr  = r + 8 * hi16;
            float rs  = __shfl(rowsum, lr, 32);
            float val = oacc[ht][r] / rs;
            int   t   = qbase + lr;
            enc[((size_t)(b * Tt + t) * Nh + n) * Hd + ht * 16 + l15] = f2bf(val);
        }
    }
}

// =====================================================================
// Kernel 4: output projection.
//   encoded [4096 x 2048] (bf16) x w_out [2048 x 2048] -> out f32.
// =====================================================================
__global__ __launch_bounds__(128) void out_proj_kernel(
    const __bf16* __restrict__ enc, const float* __restrict__ wout,
    float* __restrict__ out)
{
    __shared__ unsigned short lds_w[64 * WSTRIDE];

    const int col0 = blockIdx.x * 64;  // D column
    const int row0 = blockIdx.y * 64;  // b*T+t
    const int tid  = threadIdx.x;
    const int wave = tid >> 5;
    const int lane = tid & 31;
    const int l15  = lane & 15;
    const int hi16 = lane >> 4;
    const int kb   = 8 * hi16;
    const int kb16 = 16 * hi16;
    const int m0   = row0 + wave * 16;

    v8f acc[4];
#pragma unroll
    for (int i = 0; i < 4; ++i) acc[i] = v8f{0.f,0.f,0.f,0.f,0.f,0.f,0.f,0.f};

    const int dr = tid >> 2;
    const int cq = (tid & 3) * 16;
    const int KD = Nh * Hd; // 2048 contraction

    for (int d0 = 0; d0 < KD; d0 += 32) {
        __syncthreads();
        const float* wp = wout + (size_t)(d0 + dr) * Dd + col0 + cq;
#pragma unroll
        for (int j4 = 0; j4 < 4; ++j4) {
            float4 w4 = *(const float4*)(wp + j4 * 4);
            lds_w[(cq + j4 * 4 + 0) * WSTRIDE + dr] = f2bfbits(w4.x);
            lds_w[(cq + j4 * 4 + 1) * WSTRIDE + dr] = f2bfbits(w4.y);
            lds_w[(cq + j4 * 4 + 2) * WSTRIDE + dr] = f2bfbits(w4.z);
            lds_w[(cq + j4 * 4 + 3) * WSTRIDE + dr] = f2bfbits(w4.w);
        }
        __syncthreads();

        const __bf16* erow = enc + (size_t)(m0 + l15) * KD + d0;
        BFrag ta;
        ta.u[0] = *(const U4*)(erow + kb16);
        ta.u[1] = *(const U4*)(erow + kb16 + 8);
        v16bf af = ta.v;

#pragma unroll
        for (int ct = 0; ct < 4; ++ct) {
            v16bf bf = ld16(&lds_w[(ct * 16 + l15) * WSTRIDE + kb16]);
            acc[ct] = wmma_bf16(af, bf, acc[ct]);
        }
    }

#pragma unroll
    for (int ct = 0; ct < 4; ++ct) {
        int col = col0 + ct * 16 + l15;
#pragma unroll
        for (int r = 0; r < 8; ++r) {
            int row = m0 + r + 8 * hi16;
            out[(size_t)row * Dd + col] = acc[ct][r];
        }
    }
}

// =====================================================================
extern "C" void kernel_launch(void* const* d_in, const int* in_sizes, int n_in,
                              void* d_out, int out_size, void* d_ws, size_t ws_size,
                              hipStream_t stream) {
    (void)in_sizes; (void)n_in; (void)out_size; (void)ws_size;
    const float* x    = (const float*)d_in[0];
    // d_in[1] = positions (recomputed), d_in[2] = attn_mask (causal, implicit)
    const float* wq   = (const float*)d_in[3];
    const float* wkv  = (const float*)d_in[4];
    const float* wout = (const float*)d_in[5];
    float* out = (float*)d_out;

    // workspace layout (bf16): q | k | v_transposed | encoded  (~48 MB)
    __bf16* qw  = (__bf16*)d_ws;
    __bf16* kw  = qw + (size_t)Bb * Tt * Nh * Hd;
    __bf16* vt  = kw + (size_t)Bb * Tt * Kh * Hd;
    __bf16* enc = vt + (size_t)Bb * Tt * Kh * Hd;

    // 1) fused QKV projection (WMMA GEMM); V written transposed [b][kk][h][t]
    qkv_proj_kernel<<<dim3(64, 64), 128, 0, stream>>>(x, wq, wkv, qw, kw, vt);
    // 2) RoPE (+ H^-0.5 on q)
    {
        long long total = (long long)Bb * Tt * (Nh + Kh) * (Hd / 2);
        rope_kernel<<<(unsigned)(total / 256), 256, 0, stream>>>(qw, kw);
    }
    // 3) attention core (WMMA QK^T and PV, max-free softcapped softmax)
    attn_kernel<<<Bb * Nh * (Tt / 16), 32, 0, stream>>>(qw, kw, vt, enc);
    // 4) output projection (WMMA GEMM, f32 out)
    out_proj_kernel<<<dim3(Dd / 64, 64), 128, 0, stream>>>(enc, wout, out);
}
